// NeuralMemoryBank_80882824118732
// MI455X (gfx1250) — matrix operand order repaired and checked by reading, not compiled, not run on hardware
//
#include <hip/hip_runtime.h>
#include <hip/hip_bf16.h>
#include <math.h>

typedef __bf16 bf16_t;
typedef __bf16 v16bf __attribute__((ext_vector_type(16)));
typedef float  v8f   __attribute__((ext_vector_type(8)));
typedef int    v4i   __attribute__((ext_vector_type(4)));

#if defined(__gfx1250__) && __has_builtin(__builtin_amdgcn_global_load_async_to_lds_b128) && \
    __has_builtin(__builtin_amdgcn_s_wait_asynccnt)
#define USE_ASYNC_LDS 1
#define AS1 __attribute__((address_space(1)))
#define AS3 __attribute__((address_space(3)))
#else
#define USE_ASYNC_LDS 0
#endif

// ---------------------------------------------------------------------------
// Operand packing: fp32 row-major -> bf16 WMMA operand tiles.
// A tile (16x32): lane l holds row (l&15); elems 0..7 = K[tk*32 + (l>>4)*8 ..],
// elems 8..15 = +16.  B tile (32x16): lane l holds col (l&15), same k order
// as A.  Tile = 512 bf16; tileIdx = major*(K/32) + tk.
// ---------------------------------------------------------------------------

__global__ void pack_a_kernel(const float* __restrict__ src, bf16_t* __restrict__ dst,
                              int M, int K) {
  int tid   = blockIdx.x * blockDim.x + threadIdx.x;
  int KT    = K >> 5;
  int tiles = (M >> 4) * KT;
  int tile  = tid >> 5;
  if (tile >= tiles) return;
  int lane = tid & 31;
  int ti = tile / KT, tk = tile % KT;
  int r    = ti * 16 + (lane & 15);
  int half = lane >> 4;
  int b1   = tk * 32 + half * 8;
  const float* s = src + (size_t)r * K;
  bf16_t* d = dst + (size_t)tile * 512 + (size_t)lane * 16;
#pragma unroll
  for (int j = 0; j < 8; ++j) d[j]     = (bf16_t)s[b1 + j];
#pragma unroll
  for (int j = 0; j < 8; ++j) d[8 + j] = (bf16_t)s[b1 + 16 + j];
}

// src is [K x N] row-major
__global__ void pack_b_kernel(const float* __restrict__ src, bf16_t* __restrict__ dst,
                              int K, int Nn) {
  int tid   = blockIdx.x * blockDim.x + threadIdx.x;
  int KT    = K >> 5;
  int tiles = (Nn >> 4) * KT;
  int tile  = tid >> 5;
  if (tile >= tiles) return;
  int lane = tid & 31;
  int tn = tile / KT, tk = tile % KT;
  int c  = tn * 16 + (lane & 15);
  int kb = tk * 32 + (lane >> 4) * 8;
  bf16_t* d = dst + (size_t)tile * 512 + (size_t)lane * 16;
#pragma unroll
  for (int j = 0; j < 16; ++j) d[j] = (bf16_t)src[(size_t)(kb + j + (j >> 3) * 8) * Nn + c];
}

// src is [N x K] row-major (i.e. B^T stored row-major; contiguous gathers)
__global__ void pack_bt_kernel(const float* __restrict__ src, bf16_t* __restrict__ dst,
                               int K, int Nn) {
  int tid   = blockIdx.x * blockDim.x + threadIdx.x;
  int KT    = K >> 5;
  int tiles = (Nn >> 4) * KT;
  int tile  = tid >> 5;
  if (tile >= tiles) return;
  int lane = tid & 31;
  int tn = tile / KT, tk = tile % KT;
  int c  = tn * 16 + (lane & 15);
  int kb = tk * 32 + (lane >> 4) * 8;
  const float* s = src + (size_t)c * K + kb;
  bf16_t* d = dst + (size_t)tile * 512 + (size_t)lane * 16;
#pragma unroll
  for (int j = 0; j < 16; ++j) d[j] = (bf16_t)s[j + (j >> 3) * 8];
}

// ---------------------------------------------------------------------------
// Generic bf16 GEMM with fp32 accumulate: out = A[M x K] * B[K x N] + bias.
// 8 waves / WG; wave owns a 16x64 strip (4 D tiles).  grid = (M/16, N/512).
// Output: row-major fp32 C, or bf16 packed-A tiles (Cpack) via LDS swizzle.
// ---------------------------------------------------------------------------
__global__ __launch_bounds__(256) void gemm_bf16_kernel(
    const bf16_t* __restrict__ A, const bf16_t* __restrict__ Bm,
    const float* __restrict__ bias,
    float* __restrict__ C, bf16_t* __restrict__ Cpack,
    int M, int Nn, int K) {
  __shared__ float swz[8][16 * 64];

  int lane = threadIdx.x & 31;
  int wave = threadIdx.x >> 5;
  int ti   = blockIdx.x;
  int tnBase = (blockIdx.y * 8 + wave) * 4;
  int KT = K >> 5;

  v8f acc[4];
#pragma unroll
  for (int t = 0; t < 4; ++t) acc[t] = {};

  const bf16_t* Arow = A + ((size_t)ti * KT) * 512 + (size_t)lane * 16;
  for (int tk = 0; tk < KT; ++tk) {
    v16bf a = *reinterpret_cast<const v16bf*>(Arow + (size_t)tk * 512);
#pragma unroll
    for (int t = 0; t < 4; ++t) {
      const bf16_t* bp = Bm + ((size_t)(tnBase + t) * KT + tk) * 512 + (size_t)lane * 16;
      v16bf b = *reinterpret_cast<const v16bf*>(bp);
      acc[t] = __builtin_amdgcn_wmma_f32_16x16x32_bf16(
          false, a, false, b, (short)0, acc[t], false, false);
    }
  }

  int half = lane >> 4;
  if (C) {
#pragma unroll
    for (int t = 0; t < 4; ++t) {
      int c = (tnBase + t) * 16 + (lane & 15);
      float bv = bias ? bias[c] : 0.0f;
#pragma unroll
      for (int i = 0; i < 8; ++i) {
        int row = ti * 16 + i + 8 * half;
        C[(size_t)row * Nn + c] = acc[t][i] + bv;
      }
    }
  }
  if (Cpack) {
#pragma unroll
    for (int t = 0; t < 4; ++t) {
      int c = (tnBase + t) * 16 + (lane & 15);
      float bv = bias ? bias[c] : 0.0f;
#pragma unroll
      for (int i = 0; i < 8; ++i)
        swz[wave][(i + 8 * half) * 64 + t * 16 + (lane & 15)] = acc[t][i] + bv;
    }
    __syncthreads();
    int r   = lane & 15;
    int KTo = Nn >> 5;
#pragma unroll
    for (int ta = 0; ta < 2; ++ta) {
      size_t tileIdx = (size_t)ti * KTo + (size_t)(tnBase >> 1) + ta;
      bf16_t* d = Cpack + tileIdx * 512 + (size_t)lane * 16;
#pragma unroll
      for (int j = 0; j < 8; ++j)
        d[j]     = (bf16_t)swz[wave][r * 64 + ta * 32 + half * 8 + j];
#pragma unroll
      for (int j = 0; j < 8; ++j)
        d[8 + j] = (bf16_t)swz[wave][r * 64 + ta * 32 + 16 + half * 8 + j];
    }
  }
}

// ---------------------------------------------------------------------------
// Flash attention.  One WG = (b, h, 32 query rows = 2 row-tiles); 8 waves.
// Every K-tile / V-tile B-operand load is shared by the two row-tiles, halving
// L2 traffic vs 16-row blocks.  m-superchunks of 128: phase1 each wave does two
// 16x16 score tiles (32 WMMAs), 256-thread online softmax, P repacked to
// A-layout in LDS; phase2 each wave owns a 64-wide VD slice for both row-tiles
// (32 WMMAs).  Output emitted directly as bf16 packed-A tiles.
// ---------------------------------------------------------------------------
__global__ __launch_bounds__(256) void attn_kernel(
    const bf16_t* __restrict__ qA,   // packed A of q [4096 x 4096] (KT = 128)
    const bf16_t* __restrict__ Kb,   // packed B of K^T: tileIdx = mt*16 + tkd
    const bf16_t* __restrict__ Vb,   // packed B of V: tileIdx = tn*128 + tk
    bf16_t* __restrict__ rvA) {      // packed A of rv [4096 x 4096] (KT = 128)
  __shared__ __align__(32) bf16_t qlds[2 * 16 * 512];   // 32 KB
  __shared__ __align__(32) bf16_t plds[8 * 512];        // 8 KB
  __shared__ float slds[32 * 128];                      // 16 KB
  __shared__ float red[256];
  __shared__ float runmax[32], runsum[32], alphaS[32], newmax[32];
  __shared__ float swz[8][16 * 64];                     // 32 KB

  const float SCALE = 0.044194173824159216f;  // 1/sqrt(512)/TEMP

  int tid  = threadIdx.x;
  int lane = tid & 31;
  int wave = tid >> 5;
  int half = lane >> 4;
  int bid  = blockIdx.x;          // ((b*8 + h)*32 + nt32)
  int nt32 = bid & 31;
  int bh   = bid >> 5;
  int h    = bh & 7;
  int b    = bh >> 3;

  // Stage both row-tiles' q A-tiles (2 x 16 KB, contiguous) into LDS.
#if USE_ASYNC_LDS
  {
#pragma unroll
    for (int rt = 0; rt < 2; ++rt) {
      size_t base = ((size_t)(b * 64 + nt32 * 2 + rt) * 128 + (size_t)h * 16) * 512;
      const char* g = (const char*)(qA + base);
      char* l = (char*)&qlds[rt * 16 * 512];
#pragma unroll
      for (int k = 0; k < 4; ++k) {
        int byteOff = (tid + k * 256) * 16;
        __builtin_amdgcn_global_load_async_to_lds_b128(
            (AS1 v4i*)(g + byteOff), (AS3 v4i*)(l + byteOff), 0, 0);
      }
    }
    __builtin_amdgcn_s_wait_asynccnt(0);
  }
#else
  {
    v16bf* ls = reinterpret_cast<v16bf*>(qlds);
#pragma unroll
    for (int rt = 0; rt < 2; ++rt) {
      size_t base = ((size_t)(b * 64 + nt32 * 2 + rt) * 128 + (size_t)h * 16) * 512;
      const v16bf* gs = reinterpret_cast<const v16bf*>(qA + base);
      ls[rt * 512 + tid * 2]     = gs[tid * 2];
      ls[rt * 512 + tid * 2 + 1] = gs[tid * 2 + 1];
    }
  }
#endif
  if (tid < 32) { runmax[tid] = -3.0e38f; runsum[tid] = 0.0f; }

  v8f acc[2][4];
#pragma unroll
  for (int rt = 0; rt < 2; ++rt)
#pragma unroll
    for (int t = 0; t < 4; ++t) acc[rt][t] = {};
  __syncthreads();

  for (int mc = 0; mc < 32; ++mc) {
    // ---- phase 1: scores for m-tile (mc*8 + wave), both row-tiles ----
    int mt = mc * 8 + wave;
    v8f s0 = {}, s1 = {};
    for (int tkd = 0; tkd < 16; ++tkd) {
      v16bf bk = *reinterpret_cast<const v16bf*>(
          Kb + ((size_t)mt * 16 + tkd) * 512 + (size_t)lane * 16);
      v16bf a0 = *reinterpret_cast<const v16bf*>(&qlds[tkd * 512 + lane * 16]);
      v16bf a1 = *reinterpret_cast<const v16bf*>(&qlds[(16 + tkd) * 512 + lane * 16]);
      s0 = __builtin_amdgcn_wmma_f32_16x16x32_bf16(false, a0, false, bk, (short)0, s0, false, false);
      s1 = __builtin_amdgcn_wmma_f32_16x16x32_bf16(false, a1, false, bk, (short)0, s1, false, false);
    }
    int col = wave * 16 + (lane & 15);
#pragma unroll
    for (int i = 0; i < 8; ++i) {
      slds[(i + 8 * half) * 128 + col]      = s0[i] * SCALE;
      slds[(16 + i + 8 * half) * 128 + col] = s1[i] * SCALE;
    }
    __syncthreads();

    // ---- online softmax over this 128-wide chunk (32 rows, 256 threads) ----
    {
      int r = tid & 31, cg = tid >> 5;   // row r, 16-column group cg
      float pm = -3.0e38f;
#pragma unroll
      for (int j = 0; j < 16; ++j) pm = fmaxf(pm, slds[r * 128 + cg * 16 + j]);
      red[cg * 32 + r] = pm;
    }
    __syncthreads();
    if (tid < 32) {
      int r = tid;
      float cm = runmax[r];
      for (int c = 0; c < 8; ++c) cm = fmaxf(cm, red[c * 32 + r]);
      alphaS[r] = __expf(runmax[r] - cm);
      newmax[r] = cm;
    }
    __syncthreads();
    {
      int r = tid & 31, cg = tid >> 5;
      int rt = r >> 4, rr = r & 15;
      float nm = newmax[r];
      float ps = 0.0f;
#pragma unroll
      for (int j = 0; j < 16; ++j) {
        int jj = cg * 16 + j;
        float p = __expf(slds[r * 128 + jj] - nm);
        ps += p;
        int t = jj >> 5, k = jj & 31;
        int g2   = (k >> 3) & 1;
        int slot = (k & 7) + ((k >> 4) << 3);
        plds[(rt * 4 + t) * 512 + (rr + 16 * g2) * 16 + slot] = (bf16_t)p;
      }
      red[cg * 32 + r] = ps;
    }
    __syncthreads();
    if (tid < 32) {
      int r = tid;
      float sum = 0.0f;
      for (int c = 0; c < 8; ++c) sum += red[c * 32 + r];
      runsum[r] = runsum[r] * alphaS[r] + sum;
      runmax[r] = newmax[r];
    }
    __syncthreads();

    // ---- phase 2: rescale, then P x V-chunk; V tile shared by both rt ----
    float al8[2][8];
#pragma unroll
    for (int rt = 0; rt < 2; ++rt)
#pragma unroll
      for (int i = 0; i < 8; ++i) al8[rt][i] = alphaS[rt * 16 + i + 8 * half];
#pragma unroll
    for (int rt = 0; rt < 2; ++rt)
#pragma unroll
      for (int t = 0; t < 4; ++t)
#pragma unroll
        for (int i = 0; i < 8; ++i) acc[rt][t][i] *= al8[rt][i];

#pragma unroll
    for (int tkm = 0; tkm < 4; ++tkm) {
      v16bf ap0 = *reinterpret_cast<const v16bf*>(&plds[tkm * 512 + lane * 16]);
      v16bf ap1 = *reinterpret_cast<const v16bf*>(&plds[(4 + tkm) * 512 + lane * 16]);
#pragma unroll
      for (int t = 0; t < 4; ++t) {
        v16bf bv = *reinterpret_cast<const v16bf*>(
            Vb + ((size_t)(wave * 4 + t) * 128 + (mc * 4 + tkm)) * 512 + (size_t)lane * 16);
        acc[0][t] = __builtin_amdgcn_wmma_f32_16x16x32_bf16(false, ap0, false, bv, (short)0, acc[0][t], false, false);
        acc[1][t] = __builtin_amdgcn_wmma_f32_16x16x32_bf16(false, ap1, false, bv, (short)0, acc[1][t], false, false);
      }
    }
    __syncthreads();  // protect slds/plds before next chunk
  }

  // ---- normalize and emit packed-A bf16 tiles (cross-lane via LDS swz) ----
  int r = lane & 15;
#pragma unroll
  for (int rt = 0; rt < 2; ++rt) {
    float inv[8];
#pragma unroll
    for (int i = 0; i < 8; ++i) inv[i] = 1.0f / runsum[rt * 16 + i + 8 * half];
#pragma unroll
    for (int t = 0; t < 4; ++t)
#pragma unroll
      for (int i = 0; i < 8; ++i)
        swz[wave][(i + 8 * half) * 64 + t * 16 + (lane & 15)] = acc[rt][t][i] * inv[i];
    __syncthreads();
#pragma unroll
    for (int ta = 0; ta < 2; ++ta) {
      size_t tileIdx = (size_t)(b * 64 + nt32 * 2 + rt) * 128 +
                       (size_t)(h * 16 + wave * 2 + ta);
      bf16_t* d = rvA + tileIdx * 512 + (size_t)lane * 16;
#pragma unroll
      for (int j = 0; j < 8; ++j)
        d[j]     = (bf16_t)swz[wave][r * 64 + ta * 32 + half * 8 + j];
#pragma unroll
      for (int j = 0; j < 8; ++j)
        d[8 + j] = (bf16_t)swz[wave][r * 64 + ta * 32 + 16 + half * 8 + j];
    }
    __syncthreads();
  }
}

// ---------------------------------------------------------------------------
// Row L2-normalize: dst[r] = src[r] / max(||src[r]||, 1e-8).  One WG per row.
// ---------------------------------------------------------------------------
__global__ void normalize_rows_kernel(const float* __restrict__ src,
                                      float* __restrict__ dst, int K) {
  __shared__ float red[256];
  int row = blockIdx.x, tid = threadIdx.x;
  float ss = 0.0f;
  for (int j = tid; j < K; j += 256) { float v = src[(size_t)row * K + j]; ss += v * v; }
  red[tid] = ss; __syncthreads();
  for (int s = 128; s > 0; s >>= 1) {
    if (tid < s) red[tid] += red[tid + s];
    __syncthreads();
  }
  float rn = 1.0f / fmaxf(sqrtf(red[0]), 1e-8f);
  for (int j = tid; j < K; j += 256) dst[(size_t)row * K + j] = src[(size_t)row * K + j] * rn;
}

// Row argmax (first-max tie semantics).  One WG per row.
__global__ void argmax_rows_kernel(const float* __restrict__ sim,
                                   int* __restrict__ idx, int Nn) {
  __shared__ float bv[256];
  __shared__ int   bi[256];
  int row = blockIdx.x, tid = threadIdx.x;
  float best = -3.0e38f; int besti = 0x7fffffff;
  for (int j = tid; j < Nn; j += 256) {
    float v = sim[(size_t)row * Nn + j];
    if (v > best || (v == best && j < besti)) { best = v; besti = j; }
  }
  bv[tid] = best; bi[tid] = besti; __syncthreads();
  for (int s = 128; s > 0; s >>= 1) {
    if (tid < s) {
      if (bv[tid + s] > bv[tid] || (bv[tid + s] == bv[tid] && bi[tid + s] < bi[tid])) {
        bv[tid] = bv[tid + s]; bi[tid] = bi[tid + s];
      }
    }
    __syncthreads();
  }
  if (tid == 0) idx[row] = bi[0];
}

// ---------------------------------------------------------------------------
// Gated write: parallel over memory slots; each slot replays its matching
// writes in order (bit-exact with the reference scan: cross-slot independent).
// One WG per slot; 256 threads x 2 gate outputs.
// ---------------------------------------------------------------------------
__global__ __launch_bounds__(256) void write_update_kernel(
    const float* __restrict__ wk, const float* __restrict__ wv,
    const float* __restrict__ mk, const float* __restrict__ mv,
    const float* __restrict__ Wg, const float* __restrict__ bg,
    const int* __restrict__ idx,
    float* __restrict__ outK, float* __restrict__ outV) {
  __shared__ float sK[512], sV[512], sk[512], sv[512];
  __shared__ int sidx[1024];
  int i = blockIdx.x, tid = threadIdx.x;
  sK[tid]       = mk[(size_t)i * 512 + tid];
  sK[tid + 256] = mk[(size_t)i * 512 + tid + 256];
  sV[tid]       = mv[(size_t)i * 512 + tid];
  sV[tid + 256] = mv[(size_t)i * 512 + tid + 256];
  for (int t = tid; t < 1024; t += 256) sidx[t] = idx[t];
  __syncthreads();

  for (int t = 0; t < 1024; ++t) {
    if (sidx[t] != i) continue;  // uniform per block
    sk[tid]       = wk[(size_t)t * 512 + tid];
    sk[tid + 256] = wk[(size_t)t * 512 + tid + 256];
    sv[tid]       = wv[(size_t)t * 512 + tid];
    sv[tid + 256] = wv[(size_t)t * 512 + tid + 256];
    __syncthreads();
    float nk[2], nv[2];
#pragma unroll
    for (int u = 0; u < 2; ++u) {
      int j = tid + u * 256;
      float a = bg[j];
      for (int c = 0; c < 512; ++c) a += sk[c] * Wg[(size_t)c * 512 + j];
      for (int c = 0; c < 512; ++c) a += sV[c] * Wg[(size_t)(512 + c) * 512 + j];
      float gate = 1.0f / (1.0f + __expf(-a));
      nk[u] = gate * sk[j] + (1.0f - gate) * sK[j];
      nv[u] = gate * sv[j] + (1.0f - gate) * sV[j];
    }
    __syncthreads();
#pragma unroll
    for (int u = 0; u < 2; ++u) {
      int j = tid + u * 256;
      sK[j] = nk[u]; sV[j] = nv[u];
    }
    __syncthreads();
  }

  outK[(size_t)i * 512 + tid]       = sK[tid];
  outK[(size_t)i * 512 + tid + 256] = sK[tid + 256];
  outV[(size_t)i * 512 + tid]       = sV[tid];
  outV[(size_t)i * 512 + tid + 256] = sV[tid + 256];
}

// ---------------------------------------------------------------------------
extern "C" void kernel_launch(void* const* d_in, const int* in_sizes, int n_in,
                              void* d_out, int out_size, void* d_ws, size_t ws_size,
                              hipStream_t stream) {
  (void)in_sizes; (void)n_in; (void)out_size; (void)ws_size;
  const float* queries      = (const float*)d_in[0];
  const float* write_keys   = (const float*)d_in[1];
  const float* write_values = (const float*)d_in[2];
  const float* mem_keys     = (const float*)d_in[3];
  const float* mem_values   = (const float*)d_in[4];
  const float* Wq           = (const float*)d_in[5];
  const float* bq           = (const float*)d_in[6];
  const float* Wvp          = (const float*)d_in[7];
  const float* bvp          = (const float*)d_in[8];
  const float* Wg           = (const float*)d_in[9];
  const float* bg           = (const float*)d_in[10];

  float* read_out   = (float*)d_out;                    // [4*1024 x 512]
  float* new_keys   = read_out + (size_t)4096 * 512;    // [4096 x 512]
  float* new_values = new_keys + (size_t)4096 * 512;    // [4096 x 512]

  char* ws = (char*)d_ws;
  size_t off = 0;
  auto alloc = [&](size_t bytes) -> void* {
    void* p = ws + off;
    off += (bytes + 255) & ~(size_t)255;
    return p;
  };
  bf16_t* qryA = (bf16_t*)alloc((size_t)4096 * 512 * 2);
  bf16_t* WqB  = (bf16_t*)alloc((size_t)512 * 4096 * 2);
  bf16_t* qA   = (bf16_t*)alloc((size_t)4096 * 4096 * 2);
  bf16_t* Kb   = (bf16_t*)alloc((size_t)4096 * 512 * 2);
  bf16_t* Vb   = (bf16_t*)alloc((size_t)4096 * 512 * 2);
  bf16_t* rvA  = (bf16_t*)alloc((size_t)4096 * 4096 * 2);
  bf16_t* WvpB = (bf16_t*)alloc((size_t)4096 * 512 * 2);
  float*  knf  = (float*) alloc((size_t)1024 * 512 * 4);
  float*  mnf  = (float*) alloc((size_t)4096 * 512 * 4);
  bf16_t* knA  = (bf16_t*)alloc((size_t)1024 * 512 * 2);
  bf16_t* mnB  = (bf16_t*)alloc((size_t)4096 * 512 * 2);
  float*  simf = (float*) alloc((size_t)1024 * 4096 * 4);
  int*    idx  = (int*)   alloc((size_t)1024 * 4);

  auto packGrid = [](int M16, int K32) {  // (#row-tiles, #k-tiles)
    int thr = M16 * K32 * 32;
    return dim3((thr + 255) / 256);
  };

  // ---- read path ----
  pack_a_kernel<<<packGrid(256, 16), 256, 0, stream>>>(queries, qryA, 4096, 512);
  pack_b_kernel<<<packGrid(256, 16), 256, 0, stream>>>(Wq, WqB, 512, 4096);
  // q projection: bf16 packed-A output straight into the attention operand
  gemm_bf16_kernel<<<dim3(256, 8), 256, 0, stream>>>(
      qryA, WqB, bq, nullptr, qA, 4096, 4096, 512);
  pack_bt_kernel<<<packGrid(256, 16), 256, 0, stream>>>(mem_keys, Kb, 512, 4096);
  pack_b_kernel<<<packGrid(32, 128), 256, 0, stream>>>(mem_values, Vb, 4096, 512);
  attn_kernel<<<dim3(1024), 256, 0, stream>>>(qA, Kb, Vb, rvA);
  pack_b_kernel<<<packGrid(32, 128), 256, 0, stream>>>(Wvp, WvpB, 4096, 512);
  gemm_bf16_kernel<<<dim3(256, 1), 256, 0, stream>>>(
      rvA, WvpB, bvp, read_out, nullptr, 4096, 512, 4096);

  // ---- write path ----
  normalize_rows_kernel<<<dim3(1024), 256, 0, stream>>>(write_keys, knf, 512);
  normalize_rows_kernel<<<dim3(4096), 256, 0, stream>>>(mem_keys, mnf, 512);
  pack_a_kernel<<<packGrid(64, 16), 256, 0, stream>>>(knf, knA, 1024, 512);
  pack_bt_kernel<<<packGrid(256, 16), 256, 0, stream>>>(mnf, mnB, 512, 4096);
  gemm_bf16_kernel<<<dim3(64, 8), 256, 0, stream>>>(
      knA, mnB, nullptr, simf, nullptr, 1024, 4096, 512);
  argmax_rows_kernel<<<dim3(1024), 256, 0, stream>>>(simf, idx, 4096);
  write_update_kernel<<<dim3(4096), 256, 0, stream>>>(
      write_keys, write_values, mem_keys, mem_values, Wg, bg, idx, new_keys, new_values);
}
